// MixtureOfExperts_13211319402878
// MI455X (gfx1250) — matrix operand order repaired and checked by reading, not compiled, not run on hardware
//
#include <hip/hip_runtime.h>
#include <hip/hip_bf16.h>

#define TOKENS 8192
#define DIM    1024
#define FF     2048
#define NEXP   8
#define SLOT_CAP (TOKENS * 2 + NEXP * 128)   // 17408 padded slots max
#define MTILES   (SLOT_CAP / 128)            // 136 worst-case M tiles
#define LDA 72                               // LDS row stride (64 + 8 pad) in bf16

typedef __attribute__((ext_vector_type(16))) __bf16 v16bf;
typedef __attribute__((ext_vector_type(8)))  __bf16 v8bf;
typedef __attribute__((ext_vector_type(8)))  float  v8f;
typedef int v4i_vs __attribute__((vector_size(16)));   // builtin's pointee type

#define HAS_ASYNC_LDS __has_builtin(__builtin_amdgcn_global_load_async_to_lds_b128)

// 16-byte global -> LDS copy; async (ASYNCcnt) when the toolchain exposes it.
__device__ __forceinline__ void copy16_g2l(const __bf16* g, __bf16* l) {
#if HAS_ASYNC_LDS
    __builtin_amdgcn_global_load_async_to_lds_b128(
        (__attribute__((address_space(1))) v4i_vs*)g,
        (__attribute__((address_space(3))) v4i_vs*)l, 0, 0);
#else
    *(uint4*)l = *(const uint4*)g;
#endif
}

__device__ __forceinline__ void wait_async_all() {
#if HAS_ASYNC_LDS
#if __has_builtin(__builtin_amdgcn_s_wait_asynccnt)
    __builtin_amdgcn_s_wait_asynccnt(0);
#else
    asm volatile("s_wait_asynccnt 0x0" ::: "memory");
#endif
#endif
}

__device__ __forceinline__ float gelu_tanh(float v) {
    float z = 0.7978845608f * (v + 0.044715f * v * v * v);
#if __has_builtin(__builtin_amdgcn_tanhf)
    float th = __builtin_amdgcn_tanhf(z);
#else
    float ez = __expf(-2.0f * z);
    float th = (1.0f - ez) / (1.0f + ez);
#endif
    return 0.5f * v * (1.0f + th);
}

// ---------------- init ----------------
__global__ void k_init(int* slot_token, int* counts, int* cursor) {
    int i = blockIdx.x * 256 + threadIdx.x;
    if (i < SLOT_CAP) slot_token[i] = -1;
    if (i < NEXP) { counts[i] = 0; cursor[i] = 0; }
}

// ---------------- f32 -> bf16 (row-major, for x) ----------------
__global__ void k_cvt_bf16(const float* __restrict__ src, __bf16* __restrict__ dst, int n) {
    int i = (blockIdx.x * 256 + threadIdx.x) * 4;
    if (i >= n) return;
    float4 v = *(const float4*)(src + i);
    dst[i + 0] = (__bf16)v.x;
    dst[i + 1] = (__bf16)v.y;
    dst[i + 2] = (__bf16)v.z;
    dst[i + 3] = (__bf16)v.w;
}

// ------- f32 W[e][K][N] -> bf16 Wt[e][N][K] (tiled transpose-convert) -------
__global__ __launch_bounds__(256) void k_cvt_tr(const float* __restrict__ src,
                                                __bf16* __restrict__ dst, int K, int N) {
    __shared__ float tile[32][33];
    const int e = blockIdx.z;
    const int k0 = blockIdx.y * 32, n0 = blockIdx.x * 32;
    const int tx = threadIdx.x & 31, ty = threadIdx.x >> 5;  // 32 x 8
    const float* s = src + (size_t)e * K * N;
#pragma unroll
    for (int i = 0; i < 32; i += 8)
        tile[ty + i][tx] = s[(size_t)(k0 + ty + i) * N + n0 + tx];
    __syncthreads();
    __bf16* d = dst + (size_t)e * K * N;
#pragma unroll
    for (int i = 0; i < 32; i += 8)
        d[(size_t)(n0 + ty + i) * K + k0 + tx] = (__bf16)tile[tx][ty + i];
}

// ---------------- router: logits -> softmax -> top2 ----------------
__global__ __launch_bounds__(256) void k_router(const float* __restrict__ x,
                                                const float* __restrict__ Wr,
                                                const float* __restrict__ br,
                                                int* __restrict__ tok_eid,
                                                float* __restrict__ tok_gate,
                                                int* __restrict__ counts) {
    int wave = threadIdx.x >> 5;
    int lane = threadIdx.x & 31;
    int t = blockIdx.x * 8 + wave;
    if (t >= TOKENS) return;
    const float* xr = x + (size_t)t * DIM;
    float acc[NEXP];
#pragma unroll
    for (int e = 0; e < NEXP; ++e) acc[e] = 0.0f;
    for (int d = lane; d < DIM; d += 32) {
        float xv = xr[d];
        const float* wr = Wr + (size_t)d * NEXP;
#pragma unroll
        for (int e = 0; e < NEXP; ++e) acc[e] += xv * wr[e];
    }
#pragma unroll
    for (int e = 0; e < NEXP; ++e)
        for (int off = 16; off > 0; off >>= 1)
            acc[e] += __shfl_xor(acc[e], off, 32);
    if (lane == 0) {
        float l[NEXP], p[NEXP];
        float m = -1e30f;
#pragma unroll
        for (int e = 0; e < NEXP; ++e) { l[e] = acc[e] + br[e]; m = fmaxf(m, l[e]); }
        float s = 0.0f;
#pragma unroll
        for (int e = 0; e < NEXP; ++e) { p[e] = __expf(l[e] - m); s += p[e]; }
        float inv = 1.0f / s;
#pragma unroll
        for (int e = 0; e < NEXP; ++e) p[e] *= inv;
        int i0 = 0;
#pragma unroll
        for (int e = 1; e < NEXP; ++e) if (p[e] > p[i0]) i0 = e;
        int i1 = (i0 == 0) ? 1 : 0;
#pragma unroll
        for (int e = 0; e < NEXP; ++e) if (e != i0 && p[e] > p[i1]) i1 = e;
        tok_eid[2 * t + 0] = i0;  tok_gate[2 * t + 0] = p[i0];
        tok_eid[2 * t + 1] = i1;  tok_gate[2 * t + 1] = p[i1];
        atomicAdd(&counts[i0], 1);
        atomicAdd(&counts[i1], 1);
    }
}

// ---------------- expert bucket bases (padded to 128) ----------------
__global__ void k_bases(const int* __restrict__ counts, int* __restrict__ base,
                        int* __restrict__ padded) {
    int cum = 0;
    for (int e = 0; e < NEXP; ++e) {
        base[e] = cum;
        int pc = (counts[e] + 127) & ~127;
        padded[e] = pc;
        cum += pc;
    }
}

// ---------------- scatter tokens into expert slots ----------------
__global__ void k_scatter(const int* __restrict__ tok_eid, int* __restrict__ cursor,
                          const int* __restrict__ base, int* __restrict__ slot_token,
                          int* __restrict__ tok_slot) {
    int t = blockIdx.x * 256 + threadIdx.x;
    if (t >= TOKENS) return;
#pragma unroll
    for (int k = 0; k < 2; ++k) {
        int e = tok_eid[2 * t + k];
        int pos = atomicAdd(&cursor[e], 1);
        int slot = base[e] + pos;
        slot_token[slot] = t;
        tok_slot[2 * t + k] = slot;
    }
}

__device__ __forceinline__ int find_expert(const int* base, const int* padded, int row0) {
    for (int ee = 0; ee < NEXP; ++ee)
        if (row0 >= base[ee] && row0 < base[ee] + padded[ee]) return ee;
    return -1;
}

// =====================================================================
// Tiled bf16 WMMA GEMM body: C[128 x 64] per block, K in 64-wide stages,
// double-buffered LDS filled with async global->LDS copies.
// =====================================================================
#define GEMM_STAGE_COPY(buf, k0)                                                  \
    do {                                                                          \
        const __bf16* ag = asrc_row + (k0) + ahalf * 32;                          \
        __bf16* al = &As[buf][arow * LDA + ahalf * 32];                           \
        copy16_g2l(ag + 0, al + 0);                                               \
        copy16_g2l(ag + 8, al + 8);                                               \
        copy16_g2l(ag + 16, al + 16);                                             \
        copy16_g2l(ag + 24, al + 24);                                             \
        const __bf16* bg = bsrc_row + (k0) + bq * 16;                             \
        __bf16* bl = &Bs[buf][brow * LDA + bq * 16];                              \
        copy16_g2l(bg + 0, bl + 0);                                               \
        copy16_g2l(bg + 8, bl + 8);                                               \
    } while (0)

#define GEMM_STAGE_COMPUTE(buf)                                                   \
    do {                                                                          \
        const int m = wave * 16 + (lane & 15);                                    \
        const int kh = lane >> 4;                                                 \
        _Pragma("unroll")                                                         \
        for (int koff = 0; koff < 64; koff += 32) {                               \
            v8bf alo = *(const v8bf*)&As[buf][m * LDA + koff + kh * 8];           \
            v8bf ahi = *(const v8bf*)&As[buf][m * LDA + koff + 16 + kh * 8];      \
            v16bf a;                                                              \
            _Pragma("unroll")                                                     \
            for (int i = 0; i < 8; ++i) { a[i] = alo[i]; a[8 + i] = ahi[i]; }     \
            _Pragma("unroll")                                                     \
            for (int t4 = 0; t4 < 4; ++t4) {                                      \
                const int n = t4 * 16 + (lane & 15);                              \
                v8bf blo = *(const v8bf*)&Bs[buf][n * LDA + koff + kh * 8];       \
                v8bf bhi = *(const v8bf*)&Bs[buf][n * LDA + koff + 16 + kh * 8];  \
                v16bf b;                                                          \
                _Pragma("unroll")                                                 \
                for (int i = 0; i < 8; ++i) { b[i] = blo[i]; b[8 + i] = bhi[i]; } \
                acc[t4] = __builtin_amdgcn_wmma_f32_16x16x32_bf16(                \
                    false, a, false, b, (short)0, acc[t4], false, false);         \
            }                                                                     \
        }                                                                         \
    } while (0)

// ---------------- GEMM1: H = gelu(gather(x) @ W1t[e]^T + b1[e]) ----------------
__global__ __launch_bounds__(256) void k_gemm1(const __bf16* __restrict__ xb,
                                               const __bf16* __restrict__ w1t,
                                               const float* __restrict__ b1,
                                               const int* __restrict__ slot_token,
                                               const int* __restrict__ base,
                                               const int* __restrict__ padded,
                                               __bf16* __restrict__ Hb) {
    __shared__ __bf16 As[2][128 * LDA];
    __shared__ __bf16 Bs[2][64 * LDA];
    const int row0 = blockIdx.y * 128;
    const int e = find_expert(base, padded, row0);
    if (e < 0) return;
    const int n0 = blockIdx.x * 64;
    const int tid = threadIdx.x, lane = tid & 31, wave = tid >> 5;
    const int arow = tid >> 1, ahalf = tid & 1;   // 2 threads x 64B per A row
    const int brow = tid >> 2, bq = tid & 3;      // 4 threads x 32B per B row
    const int token = max(slot_token[row0 + arow], 0);  // padded rows: junk, never read
    const __bf16* asrc_row = xb + (size_t)token * DIM;
    const __bf16* bsrc_row = w1t + (size_t)e * DIM * FF + (size_t)(n0 + brow) * DIM;

    v8f acc[4];
#pragma unroll
    for (int i = 0; i < 4; ++i) acc[i] = (v8f)0.0f;

    GEMM_STAGE_COPY(0, 0);
    wait_async_all();
    __syncthreads();
    const int NKS = DIM / 64;
    for (int s = 0; s < NKS; ++s) {
        const int cur = s & 1;
        if (s + 1 < NKS) GEMM_STAGE_COPY(1 - cur, (s + 1) * 64);
        GEMM_STAGE_COMPUTE(cur);
        wait_async_all();
        __syncthreads();
    }

    const int colb = lane & 15, rowh = (lane >> 4) * 8;
#pragma unroll
    for (int t4 = 0; t4 < 4; ++t4) {
        const int n = n0 + t4 * 16 + colb;
        const float bias = b1[e * FF + n];
#pragma unroll
        for (int r = 0; r < 8; ++r) {
            const int m = row0 + wave * 16 + rowh + r;
            Hb[(size_t)m * FF + n] = (__bf16)gelu_tanh(acc[t4][r] + bias);
        }
    }
}

// ---------------- GEMM2: Y = H @ W2t[e]^T + b2[e] ----------------
__global__ __launch_bounds__(256) void k_gemm2(const __bf16* __restrict__ Hb,
                                               const __bf16* __restrict__ w2t,
                                               const float* __restrict__ b2,
                                               const int* __restrict__ base,
                                               const int* __restrict__ padded,
                                               float* __restrict__ Y) {
    __shared__ __bf16 As[2][128 * LDA];
    __shared__ __bf16 Bs[2][64 * LDA];
    const int row0 = blockIdx.y * 128;
    const int e = find_expert(base, padded, row0);
    if (e < 0) return;
    const int n0 = blockIdx.x * 64;
    const int tid = threadIdx.x, lane = tid & 31, wave = tid >> 5;
    const int arow = tid >> 1, ahalf = tid & 1;
    const int brow = tid >> 2, bq = tid & 3;
    const __bf16* asrc_row = Hb + (size_t)(row0 + arow) * FF;
    const __bf16* bsrc_row = w2t + (size_t)e * FF * DIM + (size_t)(n0 + brow) * FF;

    v8f acc[4];
#pragma unroll
    for (int i = 0; i < 4; ++i) acc[i] = (v8f)0.0f;

    GEMM_STAGE_COPY(0, 0);
    wait_async_all();
    __syncthreads();
    const int NKS = FF / 64;
    for (int s = 0; s < NKS; ++s) {
        const int cur = s & 1;
        if (s + 1 < NKS) GEMM_STAGE_COPY(1 - cur, (s + 1) * 64);
        GEMM_STAGE_COMPUTE(cur);
        wait_async_all();
        __syncthreads();
    }

    const int colb = lane & 15, rowh = (lane >> 4) * 8;
#pragma unroll
    for (int t4 = 0; t4 < 4; ++t4) {
        const int n = n0 + t4 * 16 + colb;
        const float bias = b2[e * DIM + n];
#pragma unroll
        for (int r = 0; r < 8; ++r) {
            const int m = row0 + wave * 16 + rowh + r;
            Y[(size_t)m * DIM + n] = acc[t4][r] + bias;
        }
    }
}

// ---------------- combine: out = g0*Y[s0] + g1*Y[s1] ----------------
__global__ void k_combine(const float* __restrict__ Y, const int* __restrict__ tok_slot,
                          const float* __restrict__ tok_gate, float* __restrict__ out) {
    const int t = blockIdx.x;
    const int c = threadIdx.x;  // 256 threads, float4 each => 1024 floats
    const int s0 = tok_slot[2 * t + 0], s1 = tok_slot[2 * t + 1];
    const float g0 = tok_gate[2 * t + 0], g1 = tok_gate[2 * t + 1];
    const float4 a = ((const float4*)(Y + (size_t)s0 * DIM))[c];
    const float4 b = ((const float4*)(Y + (size_t)s1 * DIM))[c];
    float4 r;
    r.x = g0 * a.x + g1 * b.x;
    r.y = g0 * a.y + g1 * b.y;
    r.z = g0 * a.z + g1 * b.z;
    r.w = g0 * a.w + g1 * b.w;
    ((float4*)(out + (size_t)t * DIM))[c] = r;
}

extern "C" void kernel_launch(void* const* d_in, const int* in_sizes, int n_in,
                              void* d_out, int out_size, void* d_ws, size_t ws_size,
                              hipStream_t stream) {
    const float* x  = (const float*)d_in[0];
    const float* Wr = (const float*)d_in[1];
    const float* br = (const float*)d_in[2];
    const float* W1 = (const float*)d_in[3];
    const float* b1 = (const float*)d_in[4];
    const float* W2 = (const float*)d_in[5];
    const float* b2 = (const float*)d_in[6];
    float* out = (float*)d_out;

    char* ws = (char*)d_ws;
    size_t off = 0;
    auto carve = [&](size_t bytes) { char* p = ws + off; off += (bytes + 255) & ~(size_t)255; return p; };
    __bf16* xb   = (__bf16*)carve((size_t)TOKENS * DIM * 2);
    __bf16* w1t  = (__bf16*)carve((size_t)NEXP * DIM * FF * 2);   // [e][F][D]
    __bf16* w2t  = (__bf16*)carve((size_t)NEXP * FF * DIM * 2);   // [e][D][F]
    __bf16* Hb   = (__bf16*)carve((size_t)SLOT_CAP * FF * 2);
    float*  Y    = (float*)carve((size_t)SLOT_CAP * DIM * 4);
    int*    tok_eid    = (int*)carve((size_t)TOKENS * 2 * 4);
    float*  tok_gate   = (float*)carve((size_t)TOKENS * 2 * 4);
    int*    tok_slot   = (int*)carve((size_t)TOKENS * 2 * 4);
    int*    slot_token = (int*)carve((size_t)SLOT_CAP * 4);
    int*    counts     = (int*)carve(NEXP * 4);
    int*    base       = (int*)carve(NEXP * 4);
    int*    padded     = (int*)carve(NEXP * 4);
    int*    cursor     = (int*)carve(NEXP * 4);

    k_init<<<(SLOT_CAP + 255) / 256, 256, 0, stream>>>(slot_token, counts, cursor);
    k_cvt_bf16<<<(TOKENS * DIM / 4 + 255) / 256, 256, 0, stream>>>(x, xb, TOKENS * DIM);
    k_cvt_tr<<<dim3(FF / 32, DIM / 32, NEXP), 256, 0, stream>>>(W1, w1t, DIM, FF);
    k_cvt_tr<<<dim3(DIM / 32, FF / 32, NEXP), 256, 0, stream>>>(W2, w2t, FF, DIM);
    k_router<<<TOKENS / 8, 256, 0, stream>>>(x, Wr, br, tok_eid, tok_gate, counts);
    k_bases<<<1, 1, 0, stream>>>(counts, base, padded);
    k_scatter<<<TOKENS / 256, 256, 0, stream>>>(tok_eid, cursor, base, slot_token, tok_slot);
    k_gemm1<<<dim3(FF / 64, MTILES), 256, 0, stream>>>(xb, w1t, b1, slot_token, base, padded, Hb);
    k_gemm2<<<dim3(DIM / 64, MTILES), 256, 0, stream>>>(Hb, w2t, b2, base, padded, Y);
    k_combine<<<TOKENS, 256, 0, stream>>>(Y, tok_slot, tok_gate, out);
}
